// MultiHeadedAttention_41566693491186
// MI455X (gfx1250) — compile-verified, matrix operand order checked by reading, not run on hardware
//
#include <hip/hip_runtime.h>
#include <hip/hip_bf16.h>

// ---------------------------------------------------------------------------
// Dual-attention MHA for MI455X (gfx1250, wave32, WMMA).
//   B=8, L=1024, D_MODEL=1024, H=16, Dh=64.
// Pipeline: proj_gemm (9 fused projections, fp32->bf16, V stored transposed)
//           -> dual_attn (flash softmax, S = Q1K1^T + Q2K2^T in one f32 acc)
//           -> out_gemm (O @ Wo + bo -> fp32, stream-interleaved output).
// Workspace: 9 QKV mats + 3 O mats, bf16, 8192x1024 each = 192 MiB.
// ---------------------------------------------------------------------------

typedef __attribute__((ext_vector_type(16))) __bf16 bf16x16;
typedef __attribute__((ext_vector_type(8)))  float  f32x8;

#define MROWS 8192   // B*L
#define DM    1024   // d_model
#define NH    16
#define DH    64
#define LSEQ  1024
#define NB    8

// A/B fragment loader (16-bit operands, ISA layout):
// lane = 16*half + idx; fragment element e holds k = (e<8 ? 0 : 16) + half*8 + (e&7)
// so per lane we need two contiguous 8-element (16B) chunks: [p0..p0+7], [p1..p1+7]
// with p1 = p0 + 16 whenever the k-row is contiguous in memory.
static __device__ __forceinline__ bf16x16 frag_ld(const __bf16* p0, const __bf16* p1) {
  bf16x16 f;
#pragma unroll
  for (int i = 0; i < 8; ++i) { f[i] = p0[i]; f[i + 8] = p1[i]; }
  return f;
}

static __device__ __forceinline__ f32x8 wmma_bf16(bf16x16 a, bf16x16 b, f32x8 c) {
  // D = A(16x32 bf16) * B(32x16 bf16) + C(16x16 f32)
  return __builtin_amdgcn_wmma_f32_16x16x32_bf16(false, a, false, b, (short)0, c,
                                                 false, false);
}

// Pack two fp32 into one dword of bf16 pairs (packed cvt + single b32 store).
static __device__ __forceinline__ unsigned pack_bf2(float a, float b) {
  union { __bf16 h[2]; unsigned u; } p;
  p.h[0] = (__bf16)a;
  p.h[1] = (__bf16)b;
  return p.u;
}

// ---------------------------------------------------------------------------
// Kernel 1: fused projections. blockIdx.z = w*3 + s, w in {Wq,Wk,Wv}, s stream.
// Block tile 128(M) x 64(N), K step 32, 8 waves (each 16x64 of output).
// fp32 -> bf16 conversion happens while staging tiles into LDS (packed b32).
// Q/K output layout [b][h][l][d]; V output layout [b][h][d][l] (transposed).
// ---------------------------------------------------------------------------
__global__ __launch_bounds__(256) void proj_gemm_kernel(
    const float* __restrict__ x,
    const float* __restrict__ Wq, const float* __restrict__ Wk,
    const float* __restrict__ Wv,
    const float* __restrict__ bq, const float* __restrict__ bk,
    const float* __restrict__ bv,
    __bf16* __restrict__ qkv) {
  __shared__ __bf16 Alds[128][40];  // pad 32->40 (80B rows): conflict-free b128
  __shared__ __bf16 Blds[64][40];   // stored transposed: [n][k]

  const int z = blockIdx.z;
  const int s = z % 3, w = z / 3;
  const float* W    = (w == 0) ? Wq : (w == 1) ? Wk : Wv;
  const float* bias = (w == 0) ? bq : (w == 1) ? bk : bv;
  __bf16* Y = qkv + (size_t)z * MROWS * DM;
  const bool vtrans = (w == 2);

  const int mBase = blockIdx.x * 128;
  const int nBase = blockIdx.y * 64;
  const int t = threadIdx.x;
  const int wave = t >> 5, lane = t & 31;
  const int half = lane >> 4, n16 = lane & 15;

  f32x8 acc[4];
#pragma unroll
  for (int nt = 0; nt < 4; ++nt)
#pragma unroll
    for (int r = 0; r < 8; ++r) acc[nt][r] = 0.0f;

  // A stage: thread -> 16 consecutive fp32 of one row -> 8 packed b32 stores
  const int arow = t >> 1, acol = (t & 1) * 16;
  // B stage (transpose): thread -> 2 W rows x 4 cols -> 4 packed b32 stores
  const int bk2 = (t & 15) * 2;       // even k within tile
  const int bn4 = (t >> 4) * 4;       // n group of 4

  for (int k0 = 0; k0 < DM; k0 += 32) {
    // x layout (B,L,3,DM): row stride 3*DM, stream offset s*DM
    const float* xs = x + ((size_t)(mBase + arow) * 3 + s) * DM + k0 + acol;
#pragma unroll
    for (int i = 0; i < 8; ++i)
      *(unsigned*)&Alds[arow][acol + 2 * i] = pack_bf2(xs[2 * i], xs[2 * i + 1]);

    const float* w0 = W + (size_t)(k0 + bk2) * DM + nBase + bn4;
    const float* w1 = w0 + DM;  // next k row
#pragma unroll
    for (int j = 0; j < 4; ++j)
      *(unsigned*)&Blds[bn4 + j][bk2] = pack_bf2(w0[j], w1[j]);
    __syncthreads();

    const bf16x16 a = frag_ld(&Alds[wave * 16 + n16][half * 8],
                              &Alds[wave * 16 + n16][16 + half * 8]);
#pragma unroll
    for (int nt = 0; nt < 4; ++nt) {
      const bf16x16 bf = frag_ld(&Blds[nt * 16 + n16][half * 8],
                                 &Blds[nt * 16 + n16][16 + half * 8]);
      acc[nt] = wmma_bf16(a, bf, acc[nt]);
    }
    __syncthreads();
  }

  // C layout: element r of lane -> m = r + 8*half, n = lane&15
#pragma unroll
  for (int nt = 0; nt < 4; ++nt)
#pragma unroll
    for (int r = 0; r < 8; ++r) {
      const int grow = mBase + wave * 16 + r + 8 * half;  // b*L + l
      const int gcol = nBase + nt * 16 + n16;             // h*64 + d
      const float val = acc[nt][r] + bias[gcol];
      const int bb = grow >> 10, ll = grow & 1023;
      const int hh = gcol >> 6, dd = gcol & 63;
      const size_t idx = vtrans
          ? ((size_t)(bb * NH + hh) * DH + dd) * LSEQ + ll
          : ((size_t)(bb * NH + hh) * LSEQ + ll) * DH + dd;
      Y[idx] = (__bf16)val;
    }
}

// ---------------------------------------------------------------------------
// Kernel 2: dual attention, flash-style. One wave per (stream, b, h, 16-q tile).
// Per 32-key tile: S0/S1 = Q1*K1^T + Q2*K2^T (8 WMMAs, both pairs into same
// f32 acc), scaled by 0.5/sqrt(64); online softmax with shfl_xor row reduce
// inside each 16-lane half; P staged via LDS (C-layout -> A-layout); PV via
// 4 WMMAs against pre-transposed V.
// ---------------------------------------------------------------------------
__global__ __launch_bounds__(32) void dual_attn_kernel(
    const __bf16* __restrict__ qkv, __bf16* __restrict__ O) {
  __shared__ __bf16 Plds[16][40];

  int x = blockIdx.x;
  const int qt = x & 63; x >>= 6;
  const int h  = x & 15; x >>= 4;
  const int b  = x & 7;  x >>= 3;
  const int s  = x;  // 0:H_new 1:T_new 2:R_new

  int p1, p2, vs;
  if (s == 0)      { p1 = 1; p2 = 2; vs = 0; }  // (Qt,Kt)+(Qr,Kr), V=Vh
  else if (s == 1) { p1 = 0; p2 = 2; vs = 1; }  // (Qh,Kh)+(Qr,Kr), V=Vt
  else             { p1 = 0; p2 = 1; vs = 2; }  // (Qh,Kh)+(Qt,Kt), V=Vr

  const size_t MAT = (size_t)MROWS * DM;
  const size_t headOff = (size_t)(b * NH + h) * LSEQ * DH;  // same both layouts
  const __bf16* Q1 = qkv + (size_t)p1 * MAT + headOff;        // [l][d]
  const __bf16* Q2 = qkv + (size_t)p2 * MAT + headOff;
  const __bf16* K1 = qkv + (size_t)(3 + p1) * MAT + headOff;  // [l][d]
  const __bf16* K2 = qkv + (size_t)(3 + p2) * MAT + headOff;
  const __bf16* Vt = qkv + (size_t)(6 + vs) * MAT + headOff;  // [d][l]

  const int lane = threadIdx.x;
  const int half = lane >> 4, n16 = lane & 15;
  const int qBase = qt * 16;

  // Q A-fragments for both pairs, resident for whole key loop (d=64 -> 2 ksteps)
  bf16x16 qa1[2], qa2[2];
#pragma unroll
  for (int ks = 0; ks < 2; ++ks) {
    const __bf16* p = Q1 + (size_t)(qBase + n16) * DH + ks * 32 + half * 8;
    qa1[ks] = frag_ld(p, p + 16);
    p = Q2 + (size_t)(qBase + n16) * DH + ks * 32 + half * 8;
    qa2[ks] = frag_ld(p, p + 16);
  }

  float mrow[8], lrow[8];
  f32x8 oacc[4];
#pragma unroll
  for (int r = 0; r < 8; ++r) { mrow[r] = -1e30f; lrow[r] = 0.0f; }
#pragma unroll
  for (int nt = 0; nt < 4; ++nt)
#pragma unroll
    for (int r = 0; r < 8; ++r) oacc[nt][r] = 0.0f;

  const float sscale = 0.5f / 8.0f;  // 0.5 * rsqrt(Dh=64)

  for (int k0 = 0; k0 < LSEQ; k0 += 32) {
    f32x8 s0, s1;  // keys [k0,k0+16) and [k0+16,k0+32)
#pragma unroll
    for (int r = 0; r < 8; ++r) { s0[r] = 0.0f; s1[r] = 0.0f; }

#pragma unroll
    for (int ks = 0; ks < 2; ++ks) {
      const int dOff = ks * 32 + half * 8;
      const __bf16* kp = K1 + (size_t)(k0 + n16) * DH + dOff;
      s0 = wmma_bf16(qa1[ks], frag_ld(kp, kp + 16), s0);
      kp = K2 + (size_t)(k0 + n16) * DH + dOff;
      s0 = wmma_bf16(qa2[ks], frag_ld(kp, kp + 16), s0);
      kp = K1 + (size_t)(k0 + 16 + n16) * DH + dOff;
      s1 = wmma_bf16(qa1[ks], frag_ld(kp, kp + 16), s1);
      kp = K2 + (size_t)(k0 + 16 + n16) * DH + dOff;
      s1 = wmma_bf16(qa2[ks], frag_ld(kp, kp + 16), s1);
    }
    if (k0 + 32 < LSEQ) {  // global_prefetch_b8 for the next key tile
      __builtin_prefetch(K1 + (size_t)(k0 + 32 + n16) * DH, 0, 1);
      __builtin_prefetch(K2 + (size_t)(k0 + 32 + n16) * DH, 0, 1);
    }

    float pe0[8], pe1[8];
#pragma unroll
    for (int r = 0; r < 8; ++r) {
      const float a = s0[r] * sscale, c = s1[r] * sscale;
      s0[r] = a; s1[r] = c;
      float v = fmaxf(a, c);
#pragma unroll
      for (int mm = 1; mm < 16; mm <<= 1) v = fmaxf(v, __shfl_xor(v, mm));
      const float mn = fmaxf(mrow[r], v);
      const float corr = __expf(mrow[r] - mn);
      mrow[r] = mn;
      pe0[r] = __expf(s0[r] - mn);
      pe1[r] = __expf(s1[r] - mn);
      float rs = pe0[r] + pe1[r];
#pragma unroll
      for (int mm = 1; mm < 16; mm <<= 1) rs += __shfl_xor(rs, mm);
      lrow[r] = lrow[r] * corr + rs;
#pragma unroll
      for (int nt = 0; nt < 4; ++nt) oacc[nt][r] *= corr;
    }

    // C-layout -> LDS [m][key] -> A-layout reload
#pragma unroll
    for (int r = 0; r < 8; ++r) {
      Plds[r + 8 * half][n16]      = (__bf16)pe0[r];
      Plds[r + 8 * half][16 + n16] = (__bf16)pe1[r];
    }
    __syncthreads();
    const bf16x16 pa = frag_ld(&Plds[n16][half * 8], &Plds[n16][16 + half * 8]);
#pragma unroll
    for (int nt = 0; nt < 4; ++nt) {
      // V pre-transposed [d][l]: B-frag chunks contiguous over keys
      const __bf16* vp = Vt + (size_t)(nt * 16 + n16) * LSEQ + k0 + half * 8;
      oacc[nt] = wmma_bf16(pa, frag_ld(vp, vp + 16), oacc[nt]);
    }
    __syncthreads();
  }

  // O_s row-major [b*L + l][h*64 + d], normalized by lrow
  __bf16* Os = O + (size_t)s * MAT;
#pragma unroll
  for (int nt = 0; nt < 4; ++nt)
#pragma unroll
    for (int r = 0; r < 8; ++r) {
      const int m = r + 8 * half;
      const int row = b * LSEQ + qBase + m;
      const int col = h * DH + nt * 16 + n16;
      Os[(size_t)row * DM + col] = (__bf16)(oacc[nt][r] / lrow[r]);
    }
}

// ---------------------------------------------------------------------------
// Kernel 3: output projection. Out[b][l][s][:] = O_s @ Wo + bo (fp32 out).
// Same 128x64 LDS-tiled WMMA GEMM; A staged with b128 copies, Wo converted
// on the fly with packed b32 stores.
// ---------------------------------------------------------------------------
__global__ __launch_bounds__(256) void out_gemm_kernel(
    const __bf16* __restrict__ O, const float* __restrict__ Wo,
    const float* __restrict__ bo, float* __restrict__ out) {
  __shared__ __bf16 Alds[128][40];
  __shared__ __bf16 Blds[64][40];

  const int s = blockIdx.z;
  const __bf16* A = O + (size_t)s * MROWS * DM;

  const int mBase = blockIdx.x * 128;
  const int nBase = blockIdx.y * 64;
  const int t = threadIdx.x;
  const int wave = t >> 5, lane = t & 31;
  const int half = lane >> 4, n16 = lane & 15;

  f32x8 acc[4];
#pragma unroll
  for (int nt = 0; nt < 4; ++nt)
#pragma unroll
    for (int r = 0; r < 8; ++r) acc[nt][r] = 0.0f;

  const int arow = t >> 1, acol = (t & 1) * 16;
  const int bk2 = (t & 15) * 2;
  const int bn4 = (t >> 4) * 4;

  for (int k0 = 0; k0 < DM; k0 += 32) {
    // A already bf16: two 16-byte LDS stores per thread
    const __bf16* as = A + (size_t)(mBase + arow) * DM + k0 + acol;
    *(uint4*)&Alds[arow][acol]     = *(const uint4*)&as[0];
    *(uint4*)&Alds[arow][acol + 8] = *(const uint4*)&as[8];

    const float* w0 = Wo + (size_t)(k0 + bk2) * DM + nBase + bn4;
    const float* w1 = w0 + DM;
#pragma unroll
    for (int j = 0; j < 4; ++j)
      *(unsigned*)&Blds[bn4 + j][bk2] = pack_bf2(w0[j], w1[j]);
    __syncthreads();

    const bf16x16 a = frag_ld(&Alds[wave * 16 + n16][half * 8],
                              &Alds[wave * 16 + n16][16 + half * 8]);
#pragma unroll
    for (int nt = 0; nt < 4; ++nt) {
      const bf16x16 bf = frag_ld(&Blds[nt * 16 + n16][half * 8],
                                 &Blds[nt * 16 + n16][16 + half * 8]);
      acc[nt] = wmma_bf16(a, bf, acc[nt]);
    }
    __syncthreads();
  }

#pragma unroll
  for (int nt = 0; nt < 4; ++nt)
#pragma unroll
    for (int r = 0; r < 8; ++r) {
      const int grow = mBase + wave * 16 + r + 8 * half;  // b*L + l
      const int gcol = nBase + nt * 16 + n16;
      // output layout (B, L, 3, DM)
      out[((size_t)grow * 3 + s) * DM + gcol] = acc[nt][r] + bo[gcol];
    }
}

// ---------------------------------------------------------------------------
extern "C" void kernel_launch(void* const* d_in, const int* in_sizes, int n_in,
                              void* d_out, int out_size, void* d_ws,
                              size_t ws_size, hipStream_t stream) {
  (void)in_sizes; (void)n_in; (void)out_size; (void)ws_size;
  const float* x  = (const float*)d_in[0];
  const float* Wq = (const float*)d_in[1];
  const float* bq = (const float*)d_in[2];
  const float* Wk = (const float*)d_in[3];
  const float* bk = (const float*)d_in[4];
  const float* Wv = (const float*)d_in[5];
  const float* bv = (const float*)d_in[6];
  const float* Wo = (const float*)d_in[7];
  const float* bo = (const float*)d_in[8];
  float* out = (float*)d_out;

  // Workspace: 9 QKV bf16 matrices then 3 O bf16 matrices (8192x1024 each).
  __bf16* qkv  = (__bf16*)d_ws;
  __bf16* Obuf = qkv + (size_t)9 * MROWS * DM;   // total 192 MiB used

  proj_gemm_kernel<<<dim3(MROWS / 128, DM / 64, 9), 256, 0, stream>>>(
      x, Wq, Wk, Wv, bq, bk, bv, qkv);

  // 3 streams * 8 batch * 16 heads * 64 query tiles = 24576 waves
  dual_attn_kernel<<<dim3(3 * NB * NH * (LSEQ / 16)), 32, 0, stream>>>(qkv, Obuf);

  out_gemm_kernel<<<dim3(MROWS / 128, DM / 64, 3), 256, 0, stream>>>(
      Obuf, Wo, bo, out);
}
